// BitLinear_26611617366614
// MI455X (gfx1250) — matrix operand order repaired and checked by reading, not compiled, not run on hardware
//
#include <hip/hip_runtime.h>
#include <hip/hip_bf16.h>
#include <stdint.h>

// Problem dims (fixed by the reference)
#define M_DIM 8192      // B*S = 4*2048
#define N_DIM 16384     // D_OUT
#define K_DIM 4096      // D_IN

typedef __attribute__((ext_vector_type(16))) __bf16 v16bf;
typedef __attribute__((ext_vector_type(8)))  __bf16 v8bf;
typedef __attribute__((ext_vector_type(8)))  float  v8f;
typedef __attribute__((ext_vector_type(8)))  unsigned short us8;
typedef __attribute__((ext_vector_type(4)))  int    v4i;

static __device__ __forceinline__ unsigned short f32_to_bf16_rne(float f) {
  union { float f; uint32_t u; } c; c.f = f;
  uint32_t u = c.u;
  uint32_t r = u + 0x7FFFu + ((u >> 16) & 1u);   // round-to-nearest-even
  return (unsigned short)(r >> 16);
}

// gfx1250 async DMA: global -> LDS, tracked by ASYNCcnt (no VGPR data staging)
static __device__ __forceinline__ void async_copy16(unsigned short* lds,
                                                    const unsigned short* g) {
  __builtin_amdgcn_global_load_async_to_lds_b128(
      (__attribute__((address_space(1))) v4i*)(g),
      (__attribute__((address_space(3))) v4i*)(lds),
      0, 0);
}
static __device__ __forceinline__ void wait_async0() {
  asm volatile("s_wait_asynccnt 0x0" ::: "memory");
}

// ---------------- Stage 1: deterministic two-pass mean(|W|) ----------------
__global__ void absmean_partial(const float* __restrict__ w, float* __restrict__ partials) {
  __shared__ float red[256];
  const int tid = threadIdx.x;
  const uint64_t total4 = (uint64_t)N_DIM * K_DIM / 4;       // 16,777,216 float4s
  uint64_t gid = (uint64_t)blockIdx.x * 256 + tid;
  const uint64_t stride = (uint64_t)gridDim.x * 256;
  const float4* w4 = (const float4*)w;
  float s = 0.0f;
  for (uint64_t i = gid; i < total4; i += stride) {
    float4 v = w4[i];
    s += fabsf(v.x) + fabsf(v.y) + fabsf(v.z) + fabsf(v.w);
  }
  red[tid] = s;
  __syncthreads();
  for (int off = 128; off > 0; off >>= 1) {
    if (tid < off) red[tid] += red[tid + off];
    __syncthreads();
  }
  if (tid == 0) partials[blockIdx.x] = red[0];
}

__global__ void absmean_final(const float* __restrict__ partials,
                              const float* __restrict__ scale,
                              float* __restrict__ scalars) {
  __shared__ float red[256];
  const int tid = threadIdx.x;
  float s = 0.0f;
  for (int i = tid; i < 2048; i += 256) s += partials[i];
  red[tid] = s;
  __syncthreads();
  for (int off = 128; off > 0; off >>= 1) {
    if (tid < off) red[tid] += red[tid + off];
    __syncthreads();
  }
  if (tid == 0) {
    float mean = red[0] * (1.0f / ((float)N_DIM * (float)K_DIM));
    scalars[0] = mean;               // for quantization
    scalars[1] = mean * scale[0];    // GEMM epilogue alpha
  }
}

// ---------------- Stage 2: ternary quantize W -> bf16 {-1,0,1} ----------------
__global__ void quantize_w(const float* __restrict__ w, unsigned short* __restrict__ wq,
                           const float* __restrict__ scalars) {
  const float inv = 1.0f / (scalars[0] + 1e-5f);
  const uint64_t base = ((uint64_t)blockIdx.x * 256 + threadIdx.x) * 8;
  const float4* w4 = (const float4*)(w + base);
  float4 a = w4[0], b = w4[1];
  float vv[8] = {a.x, a.y, a.z, a.w, b.x, b.y, b.z, b.w};
  us8 o;
#pragma unroll
  for (int i = 0; i < 8; ++i) {
    float t = rintf(vv[i] * inv);
    t = fminf(1.0f, fmaxf(-1.0f, t));
    o[i] = f32_to_bf16_rne(t);       // exact for {-1,0,1}
  }
  *(us8*)(wq + base) = o;
}

// ---------------- Stage 3: X fp32 -> bf16 ----------------
__global__ void convert_x(const float* __restrict__ x, unsigned short* __restrict__ xq) {
  const uint64_t base = ((uint64_t)blockIdx.x * 256 + threadIdx.x) * 8;
  const float4* x4 = (const float4*)(x + base);
  float4 a = x4[0], b = x4[1];
  float vv[8] = {a.x, a.y, a.z, a.w, b.x, b.y, b.z, b.w};
  us8 o;
#pragma unroll
  for (int i = 0; i < 8; ++i) o[i] = f32_to_bf16_rne(vv[i]);
  *(us8*)(xq + base) = o;
}

// ---------------- Stage 4: bf16 WMMA GEMM, Y = (A * B^T) * alpha ----------------
// A: [M,K] bf16 row-major (K contiguous), B: [N,K] bf16 row-major (K contiguous)
#define BM 128
#define BN 128
#define BK 64     // two WMMA K-steps per LDS stage
#define LDSS 72   // padded row stride in bf16 (144B = 36 dwords): r*36 mod 64 spans
                  // {0,4,...,60} for r=0..15 -> conflict-free B128 ds loads

__global__ void __launch_bounds__(256)
gemm_bf16_wmma(const unsigned short* __restrict__ A,
               const unsigned short* __restrict__ B,
               float* __restrict__ C,
               const float* __restrict__ scalars) {
  __shared__ unsigned short sA[2][BM * LDSS];
  __shared__ unsigned short sB[2][BN * LDSS];

  const int tid  = threadIdx.x;
  const int lane = tid & 31;
  const int wave = tid >> 5;
  const int wm   = wave >> 2;     // 0..1 : wave row (64 rows of M each)
  const int wn   = wave & 3;      // 0..3 : wave col (32 cols of N each)
  const int hi   = lane >> 4;     // lane half (ISA K-split)
  const int r16  = lane & 15;

  const int m0 = blockIdx.y * BM;
  const int n0 = blockIdx.x * BN;

  v8f acc[4][2];
#pragma unroll
  for (int i = 0; i < 4; ++i)
#pragma unroll
    for (int j = 0; j < 2; ++j)
      acc[i][j] = (v8f){0.f, 0.f, 0.f, 0.f, 0.f, 0.f, 0.f, 0.f};

  // Async DMA of one 128x64 bf16 tile pair: 128 rows * 8 chunks of 16B = 1024
  // chunks per array; 256 threads issue 4 chunks each, per array.
  auto issue_stage = [&](int buf, int kt) {
    const int k0 = kt * BK;
#pragma unroll
    for (int p = 0; p < 4; ++p) {
      const int c   = tid + p * 256;
      const int row = c >> 3;
      const int col = (c & 7) * 8;                       // bf16 elems
      const unsigned short* ga = A + (uint64_t)(m0 + row) * K_DIM + k0 + col;
      const unsigned short* gb = B + (uint64_t)(n0 + row) * K_DIM + k0 + col;
      async_copy16(&sA[buf][row * LDSS + col], ga);
      async_copy16(&sB[buf][row * LDSS + col], gb);
    }
  };

  auto compute = [&](int buf) {
    const unsigned short* sa = sA[buf];
    const unsigned short* sb = sB[buf];
#pragma unroll
    for (int ks = 0; ks < 2; ++ks) {
      const int ko = ks * 32;
      v16bf af[4];
      v16bf bfr[2];
#pragma unroll
      for (int mf = 0; mf < 4; ++mf) {
        // ISA 16-bit A layout: lanes0-15 K[0..7]|K[16..23], lanes16-31 K[8..15]|K[24..31]
        const unsigned short* base = sa + (wm * 64 + mf * 16 + r16) * LDSS + ko;
        union { v16bf v; v8bf h[2]; } u;
        u.h[0] = *(const v8bf*)(base + (hi ? 8 : 0));
        u.h[1] = *(const v8bf*)(base + 16 + (hi ? 8 : 0));
        af[mf] = u.v;
      }
#pragma unroll
      for (int nf = 0; nf < 2; ++nf) {
        // ISA 16-bit B layout: lanes0-15 K[0..7]|K[16..23], lanes16-31 K[8..15]|K[24..31]
        const unsigned short* base = sb + (wn * 32 + nf * 16 + r16) * LDSS + ko;
        union { v16bf v; v8bf h[2]; } u;
        u.h[0] = *(const v8bf*)(base + hi * 8);
        u.h[1] = *(const v8bf*)(base + 16 + hi * 8);
        bfr[nf] = u.v;
      }
#pragma unroll
      for (int mf = 0; mf < 4; ++mf)
#pragma unroll
        for (int nf = 0; nf < 2; ++nf)
          acc[mf][nf] = __builtin_amdgcn_wmma_f32_16x16x32_bf16(
              false, af[mf], false, bfr[nf], (short)0, acc[mf][nf], false, false);
    }
  };

  const int KT = K_DIM / BK;   // 64 stages
  issue_stage(0, 0);
  wait_async0();
  __syncthreads();
#pragma unroll 1
  for (int kt = 0; kt < KT; ++kt) {
    if (kt + 1 < KT) issue_stage((kt + 1) & 1, kt + 1);  // DMA overlaps WMMA
    compute(kt & 1);
    wait_async0();      // our async writes for next buffer are in LDS
    __syncthreads();    // everyone's writes visible; reads of current buf done
  }

  const float alpha = scalars[1];
  // C/D layout: VGPR j, lanes0-15 -> (M=j, N=lane), lanes16-31 -> (M=j+8, N=lane-16)
#pragma unroll
  for (int mf = 0; mf < 4; ++mf)
#pragma unroll
    for (int nf = 0; nf < 2; ++nf) {
      const int mrow = m0 + wm * 64 + mf * 16 + hi * 8;
      const int ncol = n0 + wn * 32 + nf * 16 + r16;
      float* outp = C + (uint64_t)mrow * N_DIM + ncol;
#pragma unroll
      for (int j = 0; j < 8; ++j)
        outp[(uint64_t)j * N_DIM] = acc[mf][nf][j] * alpha;
    }
}

// ---------------- Host launcher ----------------
extern "C" void kernel_launch(void* const* d_in, const int* in_sizes, int n_in,
                              void* d_out, int out_size, void* d_ws, size_t ws_size,
                              hipStream_t stream) {
  const float* x     = (const float*)d_in[0];   // [4,2048,4096]
  const float* w     = (const float*)d_in[1];   // [16384,4096]
  const float* scale = (const float*)d_in[2];   // [1]
  float* out = (float*)d_out;                   // [4,2048,16384]

  char* ws = (char*)d_ws;
  float* partials = (float*)ws;                         // 2048 floats
  float* scalars  = (float*)(ws + 8192);                // [w_mean, alpha]
  unsigned short* xq = (unsigned short*)(ws + 16384);   // M*K bf16 (64 MB)
  unsigned short* wq = xq + (uint64_t)M_DIM * K_DIM;    // N*K bf16 (128 MB)

  absmean_partial<<<2048, 256, 0, stream>>>(w, partials);
  absmean_final<<<1, 256, 0, stream>>>(partials, scale, scalars);
  quantize_w<<<(int)(((uint64_t)N_DIM * K_DIM) / 2048), 256, 0, stream>>>(w, wq, scalars);
  convert_x<<<(int)(((uint64_t)M_DIM * K_DIM) / 2048), 256, 0, stream>>>(x, xq);

  dim3 grid(N_DIM / BN, M_DIM / BM);   // 128 x 64 blocks
  gemm_bf16_wmma<<<grid, 256, 0, stream>>>(xq, wq, out, scalars);
}
